// DropEdgeGCNModel_73727408603583
// MI455X (gfx1250) — matrix operand order repaired and checked by reading, not compiled, not run on hardware
//
#include <hip/hip_runtime.h>

// ---------------------------------------------------------------------------
// 3-layer GCN (DropEdge = identity at inference) for MI455X / gfx1250.
// GEMMs use v_wmma_f32_16x16x32_f16 (f16 in, f32 accumulate).
// Edge aggregation uses vectorized gathers + non-returning global f32 atomics
// (all tables L2-resident; one wave == one edge for FO=128).
// ---------------------------------------------------------------------------

typedef __attribute__((ext_vector_type(16))) _Float16 v16h;
typedef __attribute__((ext_vector_type(8)))  _Float16 v8h;
typedef __attribute__((ext_vector_type(8)))  float    v8f;

#define BN_EPS 1e-3f

// ---------------- graph normalization ----------------

__global__ void k_deg(const int* __restrict__ dst, const float* __restrict__ ew,
                      float* __restrict__ deg, int E) {
    int e = blockIdx.x * blockDim.x + threadIdx.x;
    if (e >= E) return;
    atomicAdd(&deg[dst[e]], ew[e]);
}

__global__ void k_dinv(float* __restrict__ deg, int N) {
    int i = blockIdx.x * blockDim.x + threadIdx.x;
    if (i >= N) return;
    float d = deg[i];
    deg[i] = (d > 0.f) ? rsqrtf(d) : 0.f;
}

__global__ void k_normw(const int* __restrict__ src, const int* __restrict__ dst,
                        const float* __restrict__ ew, const float* __restrict__ dinv,
                        float* __restrict__ nw, int E) {
    int e = blockIdx.x * blockDim.x + threadIdx.x;
    if (e >= E) return;
    nw[e] = dinv[src[e]] * ew[e] * dinv[dst[e]];
}

// ---------------- precision conversion / weight packing ----------------

__global__ void k_f2h(const float* __restrict__ in, _Float16* __restrict__ out, int n) {
    int i = blockIdx.x * blockDim.x + threadIdx.x;
    if (i >= n) return;
    out[i] = (_Float16)in[i];
}

// Pack [w_gcn | w_self] (each K x Fo, row-major f32) into the per-lane WMMA
// B-fragment layout: index (kt, ct, lane) -> 16 halves with
//   n = ct*16 + lane%16,  K(j) = kt*32 + (j<8 ? (lane/16)*8 + j : 16 + (lane/16)*8 + j-8)
__global__ void k_packB(const float* __restrict__ wg, const float* __restrict__ ws,
                        _Float16* __restrict__ Bp, int K, int Fo) {
    int colTiles = (2 * Fo) >> 4;
    int total = (K >> 5) * colTiles * 32;
    int idx = blockIdx.x * blockDim.x + threadIdx.x;
    if (idx >= total) return;
    int lane = idx & 31;
    int t = idx >> 5;
    int ct = t % colTiles;
    int kt = t / colTiles;
    int n = ct * 16 + (lane & 15);
    int half = lane >> 4;
    _Float16* dstp = Bp + (size_t)idx * 16;
#pragma unroll
    for (int j = 0; j < 16; ++j) {
        int k = kt * 32 + (j < 8 ? half * 8 + j : 16 + half * 8 + (j - 8));
        float v = (n < Fo) ? wg[(size_t)k * Fo + n] : ws[(size_t)k * Fo + (n - Fo)];
        dstp[j] = (_Float16)v;
    }
}

// ---------------- WMMA GEMM: D[N x NCOL] = A[N x K] * Bpacked ----------------
// One wave computes a 16-row tile and CT adjacent 16-col tiles, reusing the
// A-fragment across the CT column tiles. All control flow is wave-uniform
// (EXEC all-ones for every v_wmma as the ISA requires).

template <int CT>
__global__ void k_gemm(const _Float16* __restrict__ A, const _Float16* __restrict__ Bp,
                       float* __restrict__ D, int rowTiles, int colGroups,
                       int K, int NCOL) {
    int wave = (int)((blockIdx.x * blockDim.x + threadIdx.x) >> 5);
    int lane = threadIdx.x & 31;
    int total = rowTiles * colGroups;
    if (wave >= total) return;

    int rt = wave / colGroups;
    int cg = wave % colGroups;
    int ktiles = K >> 5;
    int colTiles = NCOL >> 4;
    int half = lane >> 4;
    int m = rt * 16 + (lane & 15);

    v8f acc[CT];
#pragma unroll
    for (int c = 0; c < CT; ++c)
        acc[c] = (v8f){0.f, 0.f, 0.f, 0.f, 0.f, 0.f, 0.f, 0.f};

    const _Float16* arow = A + (size_t)m * K;

    for (int kt = 0; kt < ktiles; ++kt) {
        // A fragment: two contiguous b128 loads per lane
        const _Float16* ap = arow + kt * 32 + half * 8;
        v8h lo = *(const v8h*)(ap);
        v8h hi = *(const v8h*)(ap + 16);
        v16h af;
#pragma unroll
        for (int j = 0; j < 8; ++j) { af[j] = lo[j]; af[j + 8] = hi[j]; }

#pragma unroll
        for (int c = 0; c < CT; ++c) {
            int ct = cg * CT + c;
            const v16h* bp =
                (const v16h*)(Bp + ((((size_t)kt * colTiles + ct) * 32 + lane) << 4));
            v16h bf = *bp;
            acc[c] = __builtin_amdgcn_wmma_f32_16x16x32_f16(
                false, af, false, bf, (short)0, acc[c], false, false);
        }
    }

    // Store D tile: lane n = lane%16; VGPR r holds row rt*16 + r + 8*(lane/16)
#pragma unroll
    for (int c = 0; c < CT; ++c) {
        int ncol = (cg * CT + c) * 16 + (lane & 15);
#pragma unroll
        for (int r = 0; r < 8; ++r) {
            int row = rt * 16 + r + 8 * half;
            D[(size_t)row * NCOL + ncol] = acc[c][r];
        }
    }
}

// ---------------- edge scatter: agg[dst] += hw[src] * norm_w ----------------
// hs is the fused [N x 2*FO] GEMM output; columns [0,FO) are hw.
// One thread per (edge, 4-channel group): b128 gather + 4 f32 atomics.
// For FO=128 all 32 lanes of a wave share one edge -> uniform metadata loads
// and a fully contiguous 512B gather across the wave.

template <int FO>
__global__ void k_scatter(const int* __restrict__ src, const int* __restrict__ dst,
                          const float* __restrict__ nw, const float* __restrict__ hs,
                          float* __restrict__ agg, int E) {
    constexpr int GP = FO / 4;
    int idx = blockIdx.x * blockDim.x + threadIdx.x;
    if (idx >= E * GP) return;
    int e = idx / GP;
    int g = (idx - e * GP) * 4;
    float w = nw[e];
    const float4 v = *(const float4*)(hs + (size_t)src[e] * (2 * FO) + g);
    float* a = agg + (size_t)dst[e] * FO + g;
    atomicAdd(a + 0, v.x * w);   // non-returning global_atomic_add_f32
    atomicAdd(a + 1, v.y * w);
    atomicAdd(a + 2, v.z * w);
    atomicAdd(a + 3, v.w * w);
}

// ---------------- epilogue: + self + bias, BatchNorm, (ReLU), recast -------

template <int FO, bool RELU, typename OUT_T>
__global__ void k_post(const float* __restrict__ agg, const float* __restrict__ hs,
                       const float* __restrict__ bias, const float* __restrict__ gamma,
                       const float* __restrict__ beta, const float* __restrict__ mean,
                       const float* __restrict__ var, OUT_T* __restrict__ out, int N_) {
    constexpr int GP = FO / 4;
    int idx = blockIdx.x * blockDim.x + threadIdx.x;
    if (idx >= N_ * GP) return;
    int i = idx / GP;
    int g = (idx - i * GP) * 4;
    float4 a = *(const float4*)(agg + (size_t)i * FO + g);
    float4 s = *(const float4*)(hs + (size_t)i * (2 * FO) + FO + g);
    const float* ap = &a.x;
    const float* sp = &s.x;
#pragma unroll
    for (int k = 0; k < 4; ++k) {
        int c = g + k;
        float v = ap[k] + sp[k] + bias[c];
        v = (v - mean[c]) * rsqrtf(var[c] + BN_EPS) * gamma[c] + beta[c];
        if (RELU) v = fmaxf(v, 0.f);
        out[(size_t)i * FO + c] = (OUT_T)v;
    }
}

// ---------------------------------------------------------------------------

static inline int cdiv(long long a, long long b) { return (int)((a + b - 1) / b); }

extern "C" void kernel_launch(void* const* d_in, const int* in_sizes, int n_in,
                              void* d_out, int out_size, void* d_ws, size_t ws_size,
                              hipStream_t stream) {
    const int F_IN = 256, UNITS = 128, NC = 40;
    const int N = in_sizes[0] / F_IN;   // 50000
    const int E = in_sizes[2];          // 1,000,000

    const float* x  = (const float*)d_in[0];
    const int*   ei = (const int*)d_in[1];
    const float* ew = (const float*)d_in[2];
    const int* srcI = ei;
    const int* dstI = ei + E;

    const float* wg[3]  = {(const float*)d_in[3],  (const float*)d_in[10], (const float*)d_in[17]};
    const float* wsf[3] = {(const float*)d_in[4],  (const float*)d_in[11], (const float*)d_in[18]};
    const float* bb[3]  = {(const float*)d_in[5],  (const float*)d_in[12], (const float*)d_in[19]};
    const float* gm[3]  = {(const float*)d_in[6],  (const float*)d_in[13], (const float*)d_in[20]};
    const float* bt[3]  = {(const float*)d_in[7],  (const float*)d_in[14], (const float*)d_in[21]};
    const float* mn[3]  = {(const float*)d_in[8],  (const float*)d_in[15], (const float*)d_in[22]};
    const float* vr[3]  = {(const float*)d_in[9],  (const float*)d_in[16], (const float*)d_in[23]};

    // ---- workspace layout (256B aligned slices) ----
    char* wsp = (char*)d_ws;
    auto take = [&](size_t bytes) -> char* {
        char* p = wsp;
        wsp += (bytes + 255) & ~(size_t)255;
        return p;
    };
    float*    nw  = (float*)take((size_t)E * 4);
    float*    deg = (float*)take((size_t)N * 4);            // becomes dinv in place
    _Float16* xh  = (_Float16*)take((size_t)N * F_IN * 2);
    _Float16* h1  = (_Float16*)take((size_t)N * UNITS * 2);
    _Float16* h2  = (_Float16*)take((size_t)N * UNITS * 2);
    float*    hs  = (float*)take((size_t)N * 2 * UNITS * 4); // fused [hw | self], reused
    float*    agg = (float*)take((size_t)N * UNITS * 4);     // reused per layer
    _Float16* Bp1 = (_Float16*)take((size_t)(F_IN / 32) * (2 * UNITS / 16) * 32 * 16 * 2);
    _Float16* Bp2 = (_Float16*)take((size_t)(UNITS / 32) * (2 * UNITS / 16) * 32 * 16 * 2);
    _Float16* Bp3 = (_Float16*)take((size_t)(UNITS / 32) * (2 * NC / 16) * 32 * 16 * 2);

    const int TB = 256;

    // ---- symmetric GCN normalization ----
    hipMemsetAsync(deg, 0, (size_t)N * 4, stream);
    k_deg  <<<cdiv(E, TB), TB, 0, stream>>>(dstI, ew, deg, E);
    k_dinv <<<cdiv(N, TB), TB, 0, stream>>>(deg, N);
    k_normw<<<cdiv(E, TB), TB, 0, stream>>>(srcI, dstI, ew, deg, nw, E);

    // ---- f16 activations + packed fused weights ----
    k_f2h<<<cdiv((long long)N * F_IN, TB), TB, 0, stream>>>(x, xh, N * F_IN);
    k_packB<<<cdiv((F_IN / 32) * (2 * UNITS / 16) * 32, TB), TB, 0, stream>>>(wg[0], wsf[0], Bp1, F_IN, UNITS);
    k_packB<<<cdiv((UNITS / 32) * (2 * UNITS / 16) * 32, TB), TB, 0, stream>>>(wg[1], wsf[1], Bp2, UNITS, UNITS);
    k_packB<<<cdiv((UNITS / 32) * (2 * NC   / 16) * 32, TB), TB, 0, stream>>>(wg[2], wsf[2], Bp3, UNITS, NC);

    const int rowTiles = N / 16;  // 3125

    // ---- layer 1: [N,256] x [256,256] ----
    {
        int colGroups = (2 * UNITS / 16) / 4;  // 4 groups of 4 col tiles
        int waves = rowTiles * colGroups;
        k_gemm<4><<<cdiv((long long)waves * 32, TB), TB, 0, stream>>>(xh, Bp1, hs, rowTiles, colGroups, F_IN, 2 * UNITS);
        hipMemsetAsync(agg, 0, (size_t)N * UNITS * 4, stream);
        k_scatter<128><<<cdiv((long long)E * 32, TB), TB, 0, stream>>>(srcI, dstI, nw, hs, agg, E);
        k_post<128, true, _Float16><<<cdiv((long long)N * 32, TB), TB, 0, stream>>>(
            agg, hs, bb[0], gm[0], bt[0], mn[0], vr[0], h1, N);
    }

    // ---- layer 2: [N,128] x [128,256] ----
    {
        int colGroups = (2 * UNITS / 16) / 4;
        int waves = rowTiles * colGroups;
        k_gemm<4><<<cdiv((long long)waves * 32, TB), TB, 0, stream>>>(h1, Bp2, hs, rowTiles, colGroups, UNITS, 2 * UNITS);
        hipMemsetAsync(agg, 0, (size_t)N * UNITS * 4, stream);
        k_scatter<128><<<cdiv((long long)E * 32, TB), TB, 0, stream>>>(srcI, dstI, nw, hs, agg, E);
        k_post<128, true, _Float16><<<cdiv((long long)N * 32, TB), TB, 0, stream>>>(
            agg, hs, bb[1], gm[1], bt[1], mn[1], vr[1], h2, N);
    }

    // ---- layer 3: [N,128] x [128,80] -> d_out (f32) ----
    {
        int colGroups = 1;                      // 5 col tiles in one group
        int waves = rowTiles * colGroups;
        k_gemm<5><<<cdiv((long long)waves * 32, TB), TB, 0, stream>>>(h2, Bp3, hs, rowTiles, colGroups, UNITS, 2 * NC);
        hipMemsetAsync(agg, 0, (size_t)N * NC * 4, stream);
        k_scatter<40><<<cdiv((long long)E * 10, TB), TB, 0, stream>>>(srcI, dstI, nw, hs, agg, E);
        k_post<40, false, float><<<cdiv((long long)N * 10, TB), TB, 0, stream>>>(
            agg, hs, bb[2], gm[2], bt[2], mn[2], vr[2], (float*)d_out, N);
    }
}
